// SuperWarpF_87127706566755
// MI455X (gfx1250) — matrix operand order repaired and checked by reading, not compiled, not run on hardware
//
#include <hip/hip_runtime.h>

typedef __attribute__((ext_vector_type(16))) _Float16 v16h;
typedef __attribute__((ext_vector_type(8)))  _Float16 v8h;
typedef __attribute__((ext_vector_type(4)))  _Float16 v4h;
typedef __attribute__((ext_vector_type(8)))  float    v8f;

#define FILLV (-255.0f)
#define CF_RELU 1
#define CF_ADD  2
#define CF_SHUF 4

__device__ __forceinline__ v16h cat8(v8h lo, v8h hi) {
  return __builtin_shufflevector(lo, hi, 0,1,2,3,4,5,6,7,8,9,10,11,12,13,14,15);
}

// ---------------------------------------------------------------------------
// Pack OIHW f32 weights -> [cout][tap*Cin + ci] f16 (K-contiguous per cout row)
// ---------------------------------------------------------------------------
__global__ void pack_w_kernel(const float* __restrict__ w, _Float16* __restrict__ dst,
                              int Cout, int Cin) {
  int n = Cout * Cin * 9;
  int i = blockIdx.x * blockDim.x + threadIdx.x;
  if (i >= n) return;
  int co  = i / (Cin * 9);
  int r   = i % (Cin * 9);
  int tap = r / Cin;
  int ci  = r % Cin;
  dst[i] = (_Float16)w[(co * Cin + ci) * 9 + tap];
}

// ---------------------------------------------------------------------------
// Head conv: 3 planes (CHW f32) -> 64ch HWC f16, 3x3 SAME. Tiny (27 MAC/out).
// ---------------------------------------------------------------------------
__global__ void head_conv_kernel(const float* __restrict__ x, const float* __restrict__ w,
                                 const float* __restrict__ b, _Float16* __restrict__ out,
                                 int H, int W) {
  int t  = blockIdx.x * blockDim.x + threadIdx.x;
  int co = t & 63;
  int pix = t >> 6;
  if (pix >= H * W) return;
  int y = pix / W, xo = pix % W;
  float acc = b[co];
  for (int ci = 0; ci < 3; ++ci)
    for (int ky = 0; ky < 3; ++ky) {
      int iy = y + ky - 1;
      if (iy < 0 || iy >= H) continue;
      for (int kx = 0; kx < 3; ++kx) {
        int ix = xo + kx - 1;
        if (ix < 0 || ix >= W) continue;
        acc += x[(ci * H + iy) * W + ix] * w[((co * 3 + ci) * 3 + ky) * 3 + kx];
      }
    }
  out[(long)pix * 64 + co] = (_Float16)acc;
}

// ---------------------------------------------------------------------------
// Implicit-GEMM 3x3 conv, Cin=64, HWC f16 in/out, f16 WMMA compute, f32 accum.
// Block = 256 threads (8 waves), tile = 16x16 pixels x 64 cout.
// Patch staged to LDS with GLOBAL_LOAD_ASYNC_TO_LDS_B128 (ASYNCcnt tracked);
// zero padding written with plain LDS stores.
// ---------------------------------------------------------------------------
__global__ __launch_bounds__(256) void conv3x3_wmma_kernel(
    const _Float16* __restrict__ in,   // HWC64 f16
    const _Float16* __restrict__ wp,   // [Cout][576] f16 packed
    const float* __restrict__ bias,    // [Cout]
    const _Float16* __restrict__ res,  // residual HWC64 f16 (out geometry) or null
    _Float16* __restrict__ out,
    int H, int W, int flags) {
  __shared__ _Float16 sIn[18 * 18 * 64];   // 41,472 B
  const int tid = threadIdx.x;
  const int bx = blockIdx.x * 16, by = blockIdx.y * 16;
  const int co_base = blockIdx.z * 64;

  // Stage 18x18 pixel patch (128 B/pixel) as 16 B segments.
  for (int i = tid; i < 18 * 18 * 8; i += 256) {
    int seg = i & 7;                    // 16B segment (8 channels)
    int pp  = i >> 3;                   // 0..323
    int ppx = pp % 18, ppy = pp / 18;
    int ix = bx + ppx - 1, iy = by + ppy - 1;
    _Float16* lp = &sIn[pp * 64 + seg * 8];
    if (ix >= 0 && ix < W && iy >= 0 && iy < H) {
      unsigned ldsOff = (unsigned)(unsigned long long)(const void*)lp;
      unsigned long long ga =
          (unsigned long long)(const void*)&in[((long)iy * W + ix) * 64 + seg * 8];
      asm volatile("global_load_async_to_lds_b128 %0, %1, off"
                   :: "v"(ldsOff), "v"(ga) : "memory");
    } else {
      *(int4*)lp = make_int4(0, 0, 0, 0);   // zero padding
    }
  }
  // Prefetch this cout-block's weight rows (global_prefetch_b8).
  __builtin_prefetch(&wp[(size_t)(co_base + (tid & 63)) * 576], 0, 1);
  asm volatile("s_wait_asynccnt 0" ::: "memory");
  __syncthreads();

  const int lane = tid & 31;
  const int wv   = tid >> 5;     // wave 0..7 -> pixel rows 2wv, 2wv+1
  const int ml   = lane & 15;    // A: M(px) index / B: N(cout) index
  const int hf   = lane >> 4;    // lane half -> K group select

  v8f acc[2][4];
  const v8f vz = {0.f,0.f,0.f,0.f,0.f,0.f,0.f,0.f};
  for (int s = 0; s < 2; ++s)
    for (int nb = 0; nb < 4; ++nb) acc[s][nb] = vz;

  for (int kc = 0; kc < 18; ++kc) {        // K step: tap = kc/2, ci half = kc&1
    const int tap = kc >> 1;
    const int ch  = (kc & 1) * 32;
    const int ky = tap / 3, kx = tap % 3;
    v16h bf[4];
#pragma unroll
    for (int nb = 0; nb < 4; ++nb) {
      const _Float16* p = wp + (size_t)(co_base + nb * 16 + ml) * 576 + kc * 32 + hf * 16;
      bf[nb] = cat8(*(const v8h*)p, *(const v8h*)(p + 8));
    }
#pragma unroll
    for (int s = 0; s < 2; ++s) {
      const int ppy = 2 * wv + s + ky;
      const int off = (ppy * 18 + (ml + kx)) * 64 + ch + hf * 16;
      v16h af = cat8(*(const v8h*)&sIn[off], *(const v8h*)&sIn[off + 8]);
#pragma unroll
      for (int nb = 0; nb < 4; ++nb)
        acc[s][nb] = __builtin_amdgcn_wmma_f32_16x16x32_f16(
            false, af, false, bf[nb], (short)0, acc[s][nb], false, false);
    }
  }

  // Epilogue: C/D layout -> lanes 0-15: N=lane, M=r; lanes 16-31: N=lane-16, M=8+r
  for (int s = 0; s < 2; ++s) {
    const int y = by + 2 * wv + s;
    for (int nb = 0; nb < 4; ++nb) {
      const int co = co_base + nb * 16 + ml;
      const float bb = bias[co];
#pragma unroll
      for (int r = 0; r < 8; ++r) {
        const int px = hf * 8 + r;
        const int x  = bx + px;
        float v = acc[s][nb][r] + bb;
        if (flags & CF_RELU) v = fmaxf(v, 0.f);
        long idx;
        if (flags & CF_SHUF) {   // pixel shuffle r=2: co = c*4 + r1*2 + r2
          const int c = co >> 2, r1 = (co >> 1) & 1, r2 = co & 1;
          idx = ((long)(2 * y + r1) * (2 * W) + (2 * x + r2)) * 64 + c;
        } else {
          idx = ((long)y * W + x) * 64 + co;
        }
        if (flags & CF_ADD) v += (float)res[idx];
        out[idx] = (_Float16)v;
      }
    }
  }
}

// ---------------------------------------------------------------------------
// Bicubic homography warp of one 64ch HWC f16 pyramid into the 192ch f16 map.
// ---------------------------------------------------------------------------
__device__ __forceinline__ float cubicw(float t) {
  float at = fabsf(t), at2 = at * at, at3 = at2 * at;
  if (at <= 1.f) return 1.5f * at3 - 2.5f * at2 + 1.f;
  if (at <  2.f) return -0.5f * at3 + 2.5f * at2 - 4.f * at + 2.f;
  return 0.f;
}

__global__ void warp_bicubic_kernel(const _Float16* __restrict__ feat, int fh, int fw,
                                    const float* __restrict__ m, float invS,
                                    _Float16* __restrict__ dstWarp, int chOff) {
  int pix = blockIdx.x * blockDim.x + threadIdx.x;
  if (pix >= 512 * 512) return;
  int X = pix & 511, Y = pix >> 9;
  int cb = blockIdx.y * 8;

  float a00=m[0],a01=m[1],a02=m[2],a10=m[3],a11=m[4],a12=m[5],a20=m[6],a21=m[7],a22=m[8];
  float det = a00*(a11*a22-a12*a21) - a01*(a10*a22-a12*a20) + a02*(a10*a21-a11*a20);
  float id = 1.f / det;
  float i00=(a11*a22-a12*a21)*id, i01=(a02*a21-a01*a22)*id, i02=(a01*a12-a02*a11)*id;
  float i10=(a12*a20-a10*a22)*id, i11=(a00*a22-a02*a20)*id, i12=(a02*a10-a00*a12)*id;
  float i20=(a10*a21-a11*a20)*id, i21=(a01*a20-a00*a21)*id, i22=(a00*a11-a01*a10)*id;
  i00*=invS; i01*=invS; i02*=invS;
  i10*=invS; i11*=invS; i12*=invS;

  float fX = (float)X, fY = (float)Y;
  float u = i00*fX + i01*fY + i02;
  float v = i10*fX + i11*fY + i12;
  float z = i20*fX + i21*fY + i22;
  float sx = u / z, sy = v / z;

  float o[8];
  bool mask = (sx >= 0.f) && (sx <= (float)(fw - 1)) && (sy >= 0.f) && (sy <= (float)(fh - 1));
  if (!mask) {
    for (int c = 0; c < 8; ++c) o[c] = FILLV;
  } else {
    float fx = floorf(sx), fy = floorf(sy);
    float tx = sx - fx, ty = sy - fy;
    int ifx = (int)fx, ify = (int)fy;
    for (int c = 0; c < 8; ++c) o[c] = 0.f;
    for (int dy = -1; dy <= 2; ++dy) {
      float wy = cubicw(ty - (float)dy);
      int iy = min(max(ify + dy, 0), fh - 1);
      for (int dx = -1; dx <= 2; ++dx) {
        float wx = cubicw(tx - (float)dx);
        int ix = min(max(ifx + dx, 0), fw - 1);
        float wgt = wy * wx;
        v8h vv = *(const v8h*)&feat[((long)iy * fw + ix) * 64 + cb];
#pragma unroll
        for (int c = 0; c < 8; ++c) o[c] += wgt * (float)vv[c];
      }
    }
  }
  v8h ov;
#pragma unroll
  for (int c = 0; c < 8; ++c) ov[c] = (_Float16)o[c];
  *(v8h*)&dstWarp[(long)pix * 192 + chOff + cb] = ov;
}

// ---------------------------------------------------------------------------
// Recon conv: 192 -> 3, 3x3 SAME, HWC192 f16 in, NCHW f32 out. Weights in LDS.
// ---------------------------------------------------------------------------
__global__ __launch_bounds__(256) void recon_conv_kernel(
    const _Float16* __restrict__ warp, const float* __restrict__ w,
    const float* __restrict__ b, float* __restrict__ out) {
  __shared__ float sw[9 * 192 * 3];   // [tap][ci][co], 20,736 B
  for (int i = threadIdx.x; i < 3 * 192 * 9; i += 256) {
    int co = i / (192 * 9);
    int r  = i % (192 * 9);
    int ci = r / 9;
    int tap = r % 9;
    sw[(tap * 192 + ci) * 3 + co] = w[i];
  }
  __syncthreads();
  int pix = blockIdx.x * blockDim.x + threadIdx.x;
  if (pix >= 512 * 512) return;
  int X = pix & 511, Y = pix >> 9;
  float a0 = b[0], a1 = b[1], a2 = b[2];
  for (int ky = 0; ky < 3; ++ky) {
    int iy = Y + ky - 1;
    if (iy < 0 || iy >= 512) continue;
    for (int kx = 0; kx < 3; ++kx) {
      int ix = X + kx - 1;
      if (ix < 0 || ix >= 512) continue;
      const _Float16* pv = &warp[((long)iy * 512 + ix) * 192];
      const float* pw = &sw[(ky * 3 + kx) * 192 * 3];
      for (int c8 = 0; c8 < 24; ++c8) {
        v8h vv = *(const v8h*)&pv[c8 * 8];
#pragma unroll
        for (int k = 0; k < 8; ++k) {
          float f = (float)vv[k];
          int ci = c8 * 8 + k;
          a0 += f * pw[ci * 3 + 0];
          a1 += f * pw[ci * 3 + 1];
          a2 += f * pw[ci * 3 + 2];
        }
      }
    }
  }
  out[0 * 262144 + pix] = a0;
  out[1 * 262144 + pix] = a1;
  out[2 * 262144 + pix] = a2;
}

// ---------------------------------------------------------------------------
extern "C" void kernel_launch(void* const* d_in, const int* in_sizes, int n_in,
                              void* d_out, int out_size, void* d_ws, size_t ws_size,
                              hipStream_t stream) {
  (void)in_sizes; (void)n_in; (void)out_size; (void)ws_size;
  const float* x = (const float*)d_in[0];
  const float* m = (const float*)d_in[1];

  // Workspace layout (bytes); all activations f16 HWC.
  char* ws = (char*)d_ws;
  _Float16* wpack = (_Float16*)ws;                       // 1,069,056 halfs
  size_t off = 2138112;
  _Float16* F1 = (_Float16*)(ws + off); off += 2097152;  // 64@128^2 (f1 final)
  _Float16* T  = (_Float16*)(ws + off); off += 2097152;  // resblock temp
  _Float16* G  = (_Float16*)(ws + off); off += 2097152;  // trunk accumulator
  _Float16* FH = (_Float16*)(ws + off); off += 2097152;  // EDSR head/body buffer
  _Float16* F2 = (_Float16*)(ws + off); off += 8388608;  // 64@256^2 (x2 final)
  _Float16* M4 = (_Float16*)(ws + off); off += 8388608;  // 64@256^2 (x4 mid)
  _Float16* F4 = (_Float16*)(ws + off); off += 33554432; // 64@512^2 (x4 final)
  _Float16* WARP = (_Float16*)(ws + off);                // 192@512^2 (~100 MB, L2-resident)

  // f16 weight packing: 17 64->64 convs then 3 upsample convs (64->256)
  const int convIdx[17] = {4,6,8,10,12,14,16, 20,22,24,26,28, 34,36,38,40,42};
  const int uIdx[3]     = {30,44,46};
  _Float16* WP[17]; _Float16* WU[3];
  for (int i = 0; i < 17; ++i) {
    WP[i] = wpack + (size_t)i * 36864;
    pack_w_kernel<<<144, 256, 0, stream>>>((const float*)d_in[convIdx[i]], WP[i], 64, 64);
  }
  for (int j = 0; j < 3; ++j) {
    WU[j] = wpack + 626688 + (size_t)j * 147456;
    pack_w_kernel<<<576, 256, 0, stream>>>((const float*)d_in[uIdx[j]], WU[j], 256, 64);
  }

  auto conv = [&](const _Float16* in, const _Float16* wpk, int bi, const _Float16* res,
                  _Float16* outp, int H, int W, int coutBlk, int flags) {
    dim3 g(W / 16, H / 16, coutBlk);
    conv3x3_wmma_kernel<<<g, 256, 0, stream>>>(in, wpk, (const float*)d_in[bi],
                                               res, outp, H, W, flags);
  };

  // ---- x1 branch (all 128x128) ----
  head_conv_kernel<<<4096, 256, 0, stream>>>(x, (const float*)d_in[2], (const float*)d_in[3], F1, 128, 128);
  conv(F1, WP[0], 5,  nullptr, T, 128, 128, 1, CF_RELU);
  conv(T,  WP[1], 7,  F1,      G, 128, 128, 1, CF_ADD);
  conv(G,  WP[2], 9,  nullptr, T, 128, 128, 1, CF_RELU);
  conv(T,  WP[3], 11, G,       G, 128, 128, 1, CF_ADD);
  conv(G,  WP[4], 13, nullptr, T, 128, 128, 1, CF_RELU);
  conv(T,  WP[5], 15, G,       G, 128, 128, 1, CF_ADD);
  conv(G,  WP[6], 17, F1,      F1,128, 128, 1, CF_ADD);    // f1 = f + conv_t(g)

  // ---- x2 branch ----
  head_conv_kernel<<<4096, 256, 0, stream>>>(x, (const float*)d_in[18], (const float*)d_in[19], FH, 128, 128);
  conv(FH, WP[7],  21, nullptr, T, 128, 128, 1, CF_RELU);
  conv(T,  WP[8],  23, FH,      G, 128, 128, 1, CF_ADD);
  conv(G,  WP[9],  25, nullptr, T, 128, 128, 1, CF_RELU);
  conv(T,  WP[10], 27, G,       G, 128, 128, 1, CF_ADD);
  conv(G,  WP[11], 29, FH,      FH,128, 128, 1, CF_ADD);   // body + f
  conv(FH, WU[0],  31, nullptr, F2,128, 128, 4, CF_SHUF);  // upsample -> 64@256^2

  // ---- x4 branch ----
  head_conv_kernel<<<4096, 256, 0, stream>>>(x, (const float*)d_in[32], (const float*)d_in[33], FH, 128, 128);
  conv(FH, WP[12], 35, nullptr, T, 128, 128, 1, CF_RELU);
  conv(T,  WP[13], 37, FH,      G, 128, 128, 1, CF_ADD);
  conv(G,  WP[14], 39, nullptr, T, 128, 128, 1, CF_RELU);
  conv(T,  WP[15], 41, G,       G, 128, 128, 1, CF_ADD);
  conv(G,  WP[16], 43, FH,      FH,128, 128, 1, CF_ADD);
  conv(FH, WU[1],  45, nullptr, M4,128, 128, 4, CF_SHUF);  // -> 64@256^2
  conv(M4, WU[2],  47, nullptr, F4,256, 256, 4, CF_SHUF);  // -> 64@512^2

  // ---- homography warp (3 pyramids -> 192ch f16 HWC) ----
  dim3 wg(1024, 8);
  warp_bicubic_kernel<<<wg, 256, 0, stream>>>(F1, 128, 128, m, 1.f, WARP, 0);
  warp_bicubic_kernel<<<wg, 256, 0, stream>>>(F2, 256, 256, m, 2.f, WARP, 64);
  warp_bicubic_kernel<<<wg, 256, 0, stream>>>(F4, 512, 512, m, 4.f, WARP, 128);

  // ---- recon conv -> output [3,512,512] f32 ----
  recon_conv_kernel<<<1024, 256, 0, stream>>>(WARP, (const float*)d_in[48],
                                              (const float*)d_in[49], (float*)d_out);
}
